// ComposedLayer_9448928051721
// MI455X (gfx1250) — matrix operand order repaired and checked by reading, not compiled, not run on hardware
//
#include <hip/hip_runtime.h>

typedef __attribute__((ext_vector_type(16))) _Float16 v16h;
typedef __attribute__((ext_vector_type(8)))  float    v8f;
typedef __attribute__((ext_vector_type(2)))  _Float16 h2;
typedef float f4 __attribute__((ext_vector_type(4)));

#define VOCAB 50000
#define EMB   16
#define HID   32
#define BATCH 4096
#define TLEN  500
#define GATES 128   // 4*HID

#define LOG2E 1.44269504088896340736f

// Branch-free transcendentals. Native V_TANH_F32 confirmed on gfx1250.
#if __has_builtin(__builtin_amdgcn_tanhf)
__device__ __forceinline__ float fast_tanh(float x) { return __builtin_amdgcn_tanhf(x); }
__device__ __forceinline__ float fast_sigmoid(float x) {
    return fmaf(0.5f, __builtin_amdgcn_tanhf(0.5f * x), 0.5f);
}
#else
__device__ __forceinline__ float fast_tanh(float x) {
    float e = __builtin_amdgcn_exp2f((2.f * LOG2E) * x);
    return 1.f - 2.f * __builtin_amdgcn_rcpf(1.f + e);
}
__device__ __forceinline__ float fast_sigmoid(float x) {
    return __builtin_amdgcn_rcpf(1.f + __builtin_amdgcn_exp2f(-LOG2E * x));
}
#endif

// ---------------------------------------------------------------------------
// Pre-pass 1: swizzled input projection for the transposed scan:
//   xp[v*128 + half*64 + n*8 + vv] = bias[g] + emb[v]·Wk[:,g],
//   g = n*16 + half*8 + vv
// Lane (m,half) reads its 64 C-operand values as 8 pairs of contiguous b128
// loads that land directly in the WMMA C blocks. 25.6 MB -> d_ws, L2-resident.
// ---------------------------------------------------------------------------
__global__ void xproj_kernel(const float* __restrict__ emb,
                             const float* __restrict__ Wk,
                             const float* __restrict__ bias,
                             float* __restrict__ xp)
{
    const int idx = blockIdx.x * blockDim.x + threadIdx.x;
    const int j = idx & (GATES - 1);
    const int v = idx >> 7;
    if (v >= VOCAB) return;
    const int hf = j >> 6, n = (j >> 3) & 7, vv = j & 7;
    const int g  = n * 16 + hf * 8 + vv;
    const float* e = emb + (size_t)v * EMB;
    float acc = bias[g];
    #pragma unroll
    for (int k = 0; k < EMB; ++k)
        acc = fmaf(e[k], Wk[k * GATES + g], acc);
    xp[idx] = acc;
}

// ---------------------------------------------------------------------------
// Pre-pass 2: token transpose (B,T) -> (T,B): one contiguous 16 KB slab/step.
// ---------------------------------------------------------------------------
__global__ void ttrans_kernel(const int* __restrict__ in, int* __restrict__ outT)
{
    const int idx = blockIdx.x * blockDim.x + threadIdx.x;  // t*B + b
    if (idx >= BATCH * TLEN) return;
    const int b = idx & (BATCH - 1);                        // B = 2^12
    const int t = idx >> 12;
    outT[idx] = in[(size_t)b * TLEN + t];
}

// ---------------------------------------------------------------------------
// Transposed fused LSTM scan: z^T = Wr^T @ h^T, one wave = 16 batch elements.
// Lane = batch element (1 token, 1 mask per lane); VGPR index = gate row.
// A = Wr^T tiles (loop-invariant), B = h^T rebuilt per step from lane-local
// state + one half-wave __shfl_xor swap (no LDS). Pipeline: tokens prefetched
// two steps ahead, x one step ahead -> no load wait on the recurrence path.
// ---------------------------------------------------------------------------
__global__ __launch_bounds__(32) void lstm_scan_kernel(
    const int*   __restrict__ tokT,    // (T, B) transposed tokens
    const float* __restrict__ Wr,      // (HID, 128) row-major
    const float* __restrict__ xp,      // (VOCAB, 128) swizzled projection
    float*       __restrict__ out)     // (B, HID)
{
    const int L    = threadIdx.x;   // 0..31
    const int m    = L & 15;        // batch column within tile (A-row too)
    const int half = L >> 4;
    const int b0   = blockIdx.x * 16;

    // A tiles: Wr^T (gate rows x K=32 hid), contiguous K order, same as hB:
    // element e -> K = half*16+e; lane row M = m.
    v16h rA[8];
    #pragma unroll
    for (int n = 0; n < 8; ++n) {
        const int g = n * 16 + m;
        #pragma unroll
        for (int e = 0; e < 16; ++e) {
            const int K = half * 16 + e;
            rA[n][e] = (_Float16)Wr[K * GATES + g];
        }
    }

    // Per-lane state for batch element b0+m, hidden j = jt*16 + half*8 + v
    float c_s[2][8], h_s[2][8];
    #pragma unroll
    for (int jt = 0; jt < 2; ++jt)
        #pragma unroll
        for (int v = 0; v < 8; ++v) { c_s[jt][v] = 0.f; h_s[jt][v] = 0.f; }

    v16h hB;                        // h^T B-operand: element e -> K = half*16+e
    #pragma unroll
    for (int e = 0; e < 16; ++e) hB[e] = (_Float16)0.f;

    const int*  tptr  = tokT + b0 + m;
    const char* xbase = (const char*)xp + half * 256;   // half*64 floats

    auto load_x = [&](int tok_, v8f* xb) {
        const unsigned off = (unsigned)tok_ * (GATES * 4u);  // 32-bit byte off
        #pragma unroll
        for (int n = 0; n < 8; ++n) {
            union { v8f z8; f4 q[2]; } u;
            u.q[0] = *(const f4*)(xbase + off + n * 32);
            u.q[1] = *(const f4*)(xbase + off + n * 32 + 16);
            xb[n] = u.z8;
        }
    };

    // prologue: tokens for t=0,1 and x for t=0
    int tok_c = tptr[0];                               // mask/x token, step t
    int tok_n = tptr[(size_t)BATCH * 1];               // token for step t+1
    v8f xbuf[8];
    load_x(tok_c, xbuf);

    #pragma unroll 1
    for (int t = 0; t < TLEN; ++t) {
        // z = Wr^T @ h^T + x  (x was prefetched last iteration)
        v8f z[8];
        #pragma unroll
        for (int n = 0; n < 8; ++n)
            z[n] = __builtin_amdgcn_wmma_f32_16x16x32_f16(false, rA[n], false, hB,
                                                          (short)0, xbuf[n], false, false);

        // pipeline: token t+2 (issued first, consumer furthest), then x for
        // t+1 using the register-resident tok_n -> no load-wait on this path
        const float fm = (tok_c != 0) ? 1.f : 0.f;
        const int x_tok = tok_n;
        const int t2 = (t + 2 < TLEN) ? (t + 2) : (TLEN - 1);
        tok_c = tok_n;
        tok_n = tptr[(size_t)t2 * BATCH];
        load_x(x_tok, xbuf);

        // gates + cell update: one mask per lane, fma-lerp (not re-branchable)
        #pragma unroll
        for (int jt = 0; jt < 2; ++jt) {
            #pragma unroll
            for (int v = 0; v < 8; ++v) {
                const float ig = fast_sigmoid(z[0 + jt][v]);   // i : tiles 0,1
                const float fg = fast_sigmoid(z[2 + jt][v]);   // f : tiles 2,3
                const float gg = fast_tanh   (z[4 + jt][v]);   // g : tiles 4,5
                const float og = fast_sigmoid(z[6 + jt][v]);   // o : tiles 6,7
                const float cn = fg * c_s[jt][v] + ig * gg;
                const float hn = og * fast_tanh(cn);
                c_s[jt][v] = fmaf(fm, cn - c_s[jt][v], c_s[jt][v]);
                h_s[jt][v] = fmaf(fm, hn - h_s[jt][v], h_s[jt][v]);
            }
        }

        // Rebuild h^T B-operand (f16). Own j-set = {half*8+v} U {16+half*8+v};
        // B needs K = half*16 + e, so lane pairs (L, L^16) swap half of their
        // packed values: 4 dword shuffles, no LDS.
        int pk[8];
        #pragma unroll
        for (int q = 0; q < 8; ++q) {
            union { h2 h; int i; } p;
            p.h[0] = (_Float16)h_s[q >> 2][(2 * q) & 7];
            p.h[1] = (_Float16)h_s[q >> 2][(2 * q + 1) & 7];
            pk[q] = p.i;
        }
        int rcv[4];
        #pragma unroll
        for (int q = 0; q < 4; ++q) {
            const int snd = half ? pk[q] : pk[q + 4];
            rcv[q] = __shfl_xor(snd, 16, 32);
        }
        {
            union { int i[8]; v16h v; } u;
            #pragma unroll
            for (int q = 0; q < 4; ++q) {
                u.i[q]     = half ? rcv[q] : pk[q];      // K = half*16 + 0..7
                u.i[4 + q] = half ? pk[4 + q] : rcv[q];  // K = half*16 + 8..15
            }
            hB = u.v;
        }
    }

    // final h: lane owns batch b0+m, hidden {jt*16 + half*8 + 0..7}
    #pragma unroll
    for (int jt = 0; jt < 2; ++jt) {
        float* op = out + (size_t)(b0 + m) * HID + jt * 16 + half * 8;
        f4 a, b;
        #pragma unroll
        for (int v = 0; v < 4; ++v) { a[v] = h_s[jt][v]; b[v] = h_s[jt][4 + v]; }
        *(f4*)op = a;
        *(f4*)(op + 4) = b;
    }
}

extern "C" void kernel_launch(void* const* d_in, const int* in_sizes, int n_in,
                              void* d_out, int out_size, void* d_ws, size_t ws_size,
                              hipStream_t stream) {
    const int*   tokens = (const int*)  d_in[0];
    const float* emb    = (const float*)d_in[1];
    const float* Wk     = (const float*)d_in[2];
    const float* Wr     = (const float*)d_in[3];
    const float* bias   = (const float*)d_in[4];
    float* out = (float*)d_out;

    float* xproj = (float*)d_ws;                       // VOCAB*128 f32 = 25.6 MB
    int*   tokT  = (int*)((float*)d_ws + (size_t)VOCAB * GATES);  // + 8.2 MB

    {
        const int total = VOCAB * GATES;
        dim3 grid((total + 255) / 256), block(256);
        hipLaunchKernelGGL(xproj_kernel, grid, block, 0, stream, emb, Wk, bias, xproj);
    }
    {
        const int total = BATCH * TLEN;
        dim3 grid((total + 255) / 256), block(256);
        hipLaunchKernelGGL(ttrans_kernel, grid, block, 0, stream, tokens, tokT);
    }
    {
        dim3 grid(BATCH / 16), block(32);      // 256 single-wave workgroups
        hipLaunchKernelGGL(lstm_scan_kernel, grid, block, 0, stream,
                           tokT, Wr, xproj, out);
    }
}
